// MultiHeadAttention_17102559772698
// MI455X (gfx1250) — matrix operand order repaired and checked by reading, not compiled, not run on hardware
//
#include <hip/hip_runtime.h>
#include <hip/hip_bf16.h>
#include <cstdint>
#include <cstddef>

#define HIDDEN   2048
#define HEADS    16
#define HEAD_DIM 128
#define BATCH    2
#define SEQ      2048
#define MTOT     (BATCH * SEQ)   // 4096 rows

typedef __bf16 bf16_t;
typedef __attribute__((ext_vector_type(16))) __bf16 v16bf;
typedef __attribute__((ext_vector_type(8)))  __bf16 v8bf;
typedef __attribute__((ext_vector_type(8)))  float  v8f;
typedef int v4i_t __attribute__((vector_size(16)));   // pointee type of async-LDS builtins

struct U2 { uint4 a, b; };   // 32 bytes, bit-castable to v16bf

// ---------------------------------------------------------------------------
// CDNA5 async LDS DMA (GLOBAL_LOAD_ASYNC_TO_LDS_B128, ASYNCcnt) with a safe
// fallback to the proven uint4 global->VGPR->LDS path.
// Probe-derived builtin signature: (AS1 int4*, AS3 int4*, imm offset, imm cpol)
// ---------------------------------------------------------------------------
#if defined(__gfx1250__) && \
    __has_builtin(__builtin_amdgcn_global_load_async_to_lds_b128) && \
    __has_builtin(__builtin_amdgcn_s_wait_asynccnt)
#define USE_ASYNC_LDS 1
#else
#define USE_ASYNC_LDS 0
#endif

#define GLOBAL_AS __attribute__((address_space(1)))
#define LDS_AS    __attribute__((address_space(3)))

// Copy one 16-byte granule global->LDS (per-lane addresses).
__device__ inline void async_copy_granule(bf16_t* ldst, const bf16_t* gsrc) {
#if USE_ASYNC_LDS
    __builtin_amdgcn_global_load_async_to_lds_b128(
        (GLOBAL_AS v4i_t*)(v4i_t*)const_cast<bf16_t*>(gsrc),
        (LDS_AS v4i_t*)(v4i_t*)ldst, 0, 0);
#else
    *reinterpret_cast<uint4*>(ldst) = *reinterpret_cast<const uint4*>(gsrc);
#endif
}

// Wait until at most N async loads outstanding (in-order completion).
template<int N>
__device__ inline void async_wait_le() {
#if USE_ASYNC_LDS
    __builtin_amdgcn_s_wait_asynccnt(N);
#endif
}

// ---------------------------------------------------------------------------
// WMMA fragment loaders (CDNA5 §7.12.2 layouts, wave32) — each fragment is
// exactly two ds_load_b128. Strides (40/136) keep 16-B alignment.
// ---------------------------------------------------------------------------

// A-matrix 16x32 bf16, row-major [16][ld]:
// lanes 0-15: row=lane, K = 0..7 and 16..23 ; lanes 16-31: K = 8..15 and 24..31
__device__ inline v16bf load_frag_a(const bf16_t* p, int ld) {
    const int lane = threadIdx.x & 31;
    const int r  = lane & 15;
    const int kb = (lane >> 4) << 3;
    const bf16_t* q = p + r * ld + kb;
    U2 t;
    t.a = *reinterpret_cast<const uint4*>(q);
    t.b = *reinterpret_cast<const uint4*>(q + 16);
    return __builtin_bit_cast(v16bf, t);
}

// B-matrix 32x16 bf16 from N-major storage [16][ld] (i.e. B^T row-major):
// lanes 0-15: col=lane, K = 0..7 and 16..23 ; lanes 16-31: col=lane-16, K = 8..15 and 24..31
__device__ inline v16bf load_frag_b_nmajor(const bf16_t* p, int ld) {
    const int lane = threadIdx.x & 31;
    const int n  = lane & 15;
    const int kb = (lane >> 4) << 3;
    const bf16_t* q = p + n * ld + kb;
    U2 t;
    t.a = *reinterpret_cast<const uint4*>(q);
    t.b = *reinterpret_cast<const uint4*>(q + 16);
    return __builtin_bit_cast(v16bf, t);
}

// Cooperative global->LDS tile copy in 16-byte granules (used once for Q).
template<int COLS, int DST_STRIDE>
__device__ inline void copy_tile(bf16_t* __restrict__ dst, const bf16_t* __restrict__ src,
                                 int src_ld, int rows, int nthreads, int tid) {
    constexpr int G = COLS / 8;
    const int total = rows * G;
    for (int g = tid; g < total; g += nthreads) {
        const int r = g / G;
        const int c = (g - r * G) * 8;
        *reinterpret_cast<uint4*>(dst + r * DST_STRIDE + c) =
            *reinterpret_cast<const uint4*>(src + (size_t)r * src_ld + c);
    }
}

// ---------------------------------------------------------------------------
// fp32 -> bf16 conversion (vectorized)
// ---------------------------------------------------------------------------
__global__ void f32_to_bf16_kernel(const float* __restrict__ src,
                                   bf16_t* __restrict__ dst, int n) {
    int i = (blockIdx.x * blockDim.x + threadIdx.x) * 8;
    if (i + 8 <= n) {
        float4 a = *reinterpret_cast<const float4*>(src + i);
        float4 b = *reinterpret_cast<const float4*>(src + i + 4);
        bf16_t t[8] = {(bf16_t)a.x, (bf16_t)a.y, (bf16_t)a.z, (bf16_t)a.w,
                       (bf16_t)b.x, (bf16_t)b.y, (bf16_t)b.z, (bf16_t)b.w};
        *reinterpret_cast<uint4*>(dst + i) = *reinterpret_cast<uint4*>(t);
    } else {
        for (; i < n; ++i) dst[i] = (bf16_t)src[i];
    }
}

// ---------------------------------------------------------------------------
// Tiled bf16 WMMA GEMM: y[m,n] = sum_k A[m,k] * W[n,k]   (nn.Linear, x @ W^T)
// 128x128 tile / 256-thread (8-wave) WG. BK=32, async double-buffered LDS.
// MODE_Q/K: fused RoPE epilogue, bf16 out in [B,H,S,D]
// MODE_V:   bf16 out in [B,H,S,D]
// MODE_OUT: fp32 out row-major [M][HIDDEN]
// ---------------------------------------------------------------------------
enum { MODE_Q = 0, MODE_K = 1, MODE_V = 2, MODE_OUT = 3 };

template<int MODE>
__global__ __launch_bounds__(256)
void gemm_rope_kernel(const bf16_t* __restrict__ A, const bf16_t* __restrict__ W,
                      void* __restrict__ outp) {
    __shared__ __align__(16) bf16_t sA[2][128 * 40];   // 128x32 (+8 pad), double buffer
    __shared__ __align__(16) bf16_t sB[2][128 * 40];

    const int tid  = threadIdx.x;
    const int lane = tid & 31;
    const int wave = tid >> 5;
    const int ln   = lane & 15;
    const int hi   = lane >> 4;          // C-layout: M = r + hi*8
    const int tileN = blockIdx.x;        // 0..15  (one head per tile for Q/K/V)
    const int tileM = blockIdx.y;        // 0..31

    const bf16_t* Ab = A + (size_t)tileM * 128 * HIDDEN;
    const bf16_t* Wb = W + (size_t)tileN * 128 * HIDDEN;

    // 128x32 tile = 512 16B-granules; 2 per thread per matrix (4 async/iter)
    const int gr0 = tid >> 2, gr1 = (tid + 256) >> 2;
    const int gc  = (tid & 3) * 8;

    auto stage = [&](int buf, const bf16_t* a, const bf16_t* w) {
        async_copy_granule(&sA[buf][gr0 * 40 + gc], a + (size_t)gr0 * HIDDEN + gc);
        async_copy_granule(&sB[buf][gr0 * 40 + gc], w + (size_t)gr0 * HIDDEN + gc);
        async_copy_granule(&sA[buf][gr1 * 40 + gc], a + (size_t)gr1 * HIDDEN + gc);
        async_copy_granule(&sB[buf][gr1 * 40 + gc], w + (size_t)gr1 * HIDDEN + gc);
    };

    v8f acc[8] = {};
    stage(0, Ab, Wb);

    for (int k0 = 0; k0 < HIDDEN; k0 += 32) {
        const int cur = (k0 >> 5) & 1;
        __syncthreads();                       // prev compute done: cur^1 reusable
        if (k0 + 32 < HIDDEN) {                // launch next tile DMA
            stage(cur ^ 1, Ab + k0 + 32, Wb + k0 + 32);
            async_wait_le<4>();                // cur's 4 loads done; next in flight
        } else {
            async_wait_le<0>();
        }
        __syncthreads();                       // publish cur tile
        v16bf af = load_frag_a(sA[cur] + (wave * 16) * 40, 40);
#pragma unroll
        for (int t = 0; t < 8; ++t) {
            v16bf bf = load_frag_b_nmajor(sB[cur] + (t * 16) * 40, 40);
            acc[t] = __builtin_amdgcn_wmma_f32_16x16x32_bf16(
                false, af, false, bf, (short)0, acc[t], false, false);
        }
    }

    if (MODE == MODE_OUT) {
        float* out = (float*)outp;
#pragma unroll
        for (int r = 0; r < 8; ++r) {
            const int m = tileM * 128 + wave * 16 + r + hi * 8;
            float* orow = out + (size_t)m * HIDDEN + tileN * 128 + ln;
#pragma unroll
            for (int t = 0; t < 8; ++t) orow[t * 16] = acc[t][r];
        }
    } else if (MODE == MODE_V) {
        bf16_t* out = (bf16_t*)outp;
        const int h = tileN;
#pragma unroll
        for (int r = 0; r < 8; ++r) {
            const int m = tileM * 128 + wave * 16 + r + hi * 8;
            const int s = m & (SEQ - 1);
            const int b = m >> 11;
            bf16_t* orow = out + ((size_t)(b * HEADS + h) * SEQ + s) * HEAD_DIM + ln;
#pragma unroll
            for (int t = 0; t < 8; ++t) orow[t * 16] = (bf16_t)acc[t][r];
        }
    } else {  // MODE_Q / MODE_K : fused RoPE. d and d+64 are tiles t and t+4,
              // same lane & accumulator slot -> rotation is register-local.
        bf16_t* out = (bf16_t*)outp;
        const int h = tileN;
        float invf[4];
#pragma unroll
        for (int t = 0; t < 4; ++t)   // inv_freq(d) = 10000^(-d/64), d = t*16+ln < 64
            invf[t] = __expf(-(float)(t * 16 + ln) * 0.14391157f);  // ln(1e4)/64
#pragma unroll
        for (int r = 0; r < 8; ++r) {
            const int m = tileM * 128 + wave * 16 + r + hi * 8;
            const int s = m & (SEQ - 1);
            const int b = m >> 11;
            const float sf = (float)s;
            bf16_t* orow = out + ((size_t)(b * HEADS + h) * SEQ + s) * HEAD_DIM + ln;
#pragma unroll
            for (int t = 0; t < 4; ++t) {
                float sn, cs;
                __sincosf(sf * invf[t], &sn, &cs);
                const float x1 = acc[t][r];
                const float x2 = acc[t + 4][r];
                orow[t * 16]      = (bf16_t)(x1 * cs - x2 * sn);
                orow[t * 16 + 64] = (bf16_t)(x2 * cs + x1 * sn);
            }
        }
    }
}

// ---------------------------------------------------------------------------
// Flash attention: one (b,h,64-row q-block) per 128-thread (4-wave) WG.
// Online softmax over K/V tiles of 32; causal mask analytic.
// K tiles: async double-buffered LDS DMA. V tiles: register path with
// transpose-on-LDS-store so PV B-fragments are contiguous b128 loads.
// ---------------------------------------------------------------------------
__global__ __launch_bounds__(128)
void attention_kernel(const bf16_t* __restrict__ Q, const bf16_t* __restrict__ K,
                      const bf16_t* __restrict__ V, bf16_t* __restrict__ Out) {
    __shared__ __align__(16) bf16_t sQ [64 * 136];     // 64 q-rows x 128 d (+8 pad)
    __shared__ __align__(16) bf16_t sK [2][32 * 136];  // double-buffered K tiles
    __shared__ __align__(16) bf16_t sVt[128 * 40];     // V^T: [d][s] (+8 pad)
    __shared__ __align__(16) bf16_t sP [64 * 40];      // probs 64 x 32 (+8 pad)
    __shared__ float sS[64 * 33];                      // raw scores (+1 pad)
    __shared__ float sF[64];                           // per-row rescale / 1/l

    const int tid  = threadIdx.x;
    const int lane = tid & 31;
    const int wave = tid >> 5;           // 4 waves -> rows wave*16 .. +15
    const int ln   = lane & 15;
    const int hi   = lane >> 4;
    const int qb   = blockIdx.x * 64;    // q-block start
    const int bh   = blockIdx.y;         // b*HEADS + h
    const int b    = bh >> 4;
    const int h    = bh & 15;

    const bf16_t* Qg = Q + (size_t)bh * SEQ * HEAD_DIM;
    const bf16_t* Kg = K + (size_t)bh * SEQ * HEAD_DIM;
    const bf16_t* Vg = V + (size_t)bh * SEQ * HEAD_DIM;

    copy_tile<128, 136>(sQ, Qg + (size_t)qb * HEAD_DIM, HEAD_DIM, 64, 128, tid);

    // 32x128 K/V tile = 512 16B-granules; 4 per thread
    const int krow[4] = { (tid + 0)   >> 4, (tid + 128) >> 4,
                          (tid + 256) >> 4, (tid + 384) >> 4 };
    const int kcol = (tid & 15) * 8;

    auto stage_k = [&](int buf, int kb) {   // async DMA, 4 granules/thread
#pragma unroll
        for (int i = 0; i < 4; ++i)
            async_copy_granule(&sK[buf][krow[i] * 136 + kcol],
                               Kg + (size_t)(kb + krow[i]) * HEAD_DIM + kcol);
    };

    uint4 rv[4];
    auto gload_v = [&](int kb) {
#pragma unroll
        for (int i = 0; i < 4; ++i)
            rv[i] = *reinterpret_cast<const uint4*>(
                Vg + (size_t)(kb + krow[i]) * HEAD_DIM + kcol);
    };
    auto store_vt = [&]() {                 // transpose on LDS store
#pragma unroll
        for (int i = 0; i < 4; ++i) {
            v8bf w = __builtin_bit_cast(v8bf, rv[i]);
#pragma unroll
            for (int j = 0; j < 8; ++j) sVt[(kcol + j) * 40 + krow[i]] = w[j];
        }
    };

    float m_i = -3.0e38f, l_i = 0.0f;    // row stats (threads 0..63 own a row)
    v8f o[8] = {};                       // 16 rows x 128 d per wave

    stage_k(0, 0);
    gload_v(0);

    const int kend = qb + 64;
    for (int kb0 = 0; kb0 < kend; kb0 += 32) {
        const int cur = (kb0 >> 5) & 1;
        __syncthreads();                 // prev PV done: sVt / sK[cur^1] reusable
        store_vt();
        if (kb0 + 32 < kend) {
            stage_k(cur ^ 1, kb0 + 32);  // next K tile DMA in flight
            gload_v(kb0 + 32);
            async_wait_le<4>();          // current K tile landed
        } else {
            async_wait_le<0>();
        }
        __syncthreads();                 // publish sK[cur] + sVt

        // scores: 16 q-rows x 32 k-cols, contraction over d=128
        v8f sc[2] = {};
#pragma unroll
        for (int kk = 0; kk < 128; kk += 32) {
            v16bf af = load_frag_a(sQ + (wave * 16) * 136 + kk, 136);
#pragma unroll
            for (int t = 0; t < 2; ++t) {
                v16bf bf = load_frag_b_nmajor(sK[cur] + (t * 16) * 136 + kk, 136);
                sc[t] = __builtin_amdgcn_wmma_f32_16x16x32_bf16(
                    false, af, false, bf, (short)0, sc[t], false, false);
            }
        }
        // scale + causal mask -> LDS
#pragma unroll
        for (int t = 0; t < 2; ++t) {
#pragma unroll
            for (int r = 0; r < 8; ++r) {
                const int lrow = wave * 16 + r + hi * 8;
                const int col  = kb0 + t * 16 + ln;
                float v = sc[t][r] * 0.08838834764f;   // 1/sqrt(128)
                if (col > qb + lrow) v = -10000.0f;    // reference mask value
                sS[lrow * 33 + t * 16 + ln] = v;
            }
        }
        __syncthreads();

        // online softmax: one thread per row
        if (tid < 64) {
            const float* row = sS + tid * 33;
            float mx = row[0];
#pragma unroll
            for (int j = 1; j < 32; ++j) mx = fmaxf(mx, row[j]);
            const float mnew = fmaxf(m_i, mx);
            const float rs = __expf(m_i - mnew);
            float sum = 0.0f;
            bf16_t* prow = sP + tid * 40;
#pragma unroll
            for (int j = 0; j < 32; ++j) {
                const float p = __expf(row[j] - mnew);
                prow[j] = (bf16_t)p;
                sum += p;
            }
            l_i = l_i * rs + sum;
            m_i = mnew;
            sF[tid] = rs;
        }
        __syncthreads();

        // rescale running output, then O += P (16x32) * V (32x128)
#pragma unroll
        for (int r = 0; r < 8; ++r) {
            const float fr = sF[wave * 16 + r + hi * 8];
#pragma unroll
            for (int t = 0; t < 8; ++t) o[t][r] *= fr;
        }
        v16bf pa = load_frag_a(sP + (wave * 16) * 40, 40);
#pragma unroll
        for (int t = 0; t < 8; ++t) {
            v16bf vb = load_frag_b_nmajor(sVt + (t * 16) * 40, 40);
            o[t] = __builtin_amdgcn_wmma_f32_16x16x32_bf16(
                false, pa, false, vb, (short)0, o[t], false, false);
        }
    }

    __syncthreads();
    if (tid < 64) sF[tid] = 1.0f / l_i;
    __syncthreads();

    // write [B,S,HIDDEN] bf16 (head h occupies cols h*128..h*128+127)
#pragma unroll
    for (int r = 0; r < 8; ++r) {
        const int lrow = wave * 16 + r + hi * 8;
        const float inv = sF[lrow];
        bf16_t* orow = Out + ((size_t)(b * SEQ + qb + lrow)) * HIDDEN + h * HEAD_DIM + ln;
#pragma unroll
        for (int t = 0; t < 8; ++t) orow[t * 16] = (bf16_t)(o[t][r] * inv);
    }
}

// ---------------------------------------------------------------------------
// Launch
// ---------------------------------------------------------------------------
extern "C" void kernel_launch(void* const* d_in, const int* in_sizes, int n_in,
                              void* d_out, int out_size, void* d_ws, size_t ws_size,
                              hipStream_t stream) {
    (void)in_sizes; (void)n_in; (void)out_size; (void)ws_size;
    const float* x  = (const float*)d_in[0];
    const float* Wq = (const float*)d_in[1];
    const float* Wk = (const float*)d_in[2];
    const float* Wv = (const float*)d_in[3];
    const float* Wo = (const float*)d_in[4];
    // d_in[5]: causal mask -- known tril, applied analytically in-kernel.

    char* ws = (char*)d_ws;
    size_t off = 0;
    auto carve = [&](size_t bytes) {
        char* p = ws + off;
        off += (bytes + 255) & ~(size_t)255;
        return p;
    };
    const size_t actB = (size_t)MTOT * HIDDEN * sizeof(bf16_t);   // 16 MiB
    const size_t wB   = (size_t)HIDDEN * HIDDEN * sizeof(bf16_t); //  8 MiB
    bf16_t* Xb  = (bf16_t*)carve(actB);
    bf16_t* Wqb = (bf16_t*)carve(wB);
    bf16_t* Wkb = (bf16_t*)carve(wB);
    bf16_t* Wvb = (bf16_t*)carve(wB);
    bf16_t* Wob = (bf16_t*)carve(wB);
    bf16_t* Qg  = (bf16_t*)carve(actB);
    bf16_t* Kg  = (bf16_t*)carve(actB);
    bf16_t* Vg  = (bf16_t*)carve(actB);
    bf16_t* AOb = (bf16_t*)carve(actB);

    const int nAct = MTOT * HIDDEN;       // 8,388,608
    const int nW   = HIDDEN * HIDDEN;     // 4,194,304
    auto cgrid = [](int n) { return dim3((unsigned)((n / 8 + 255) / 256)); };
    f32_to_bf16_kernel<<<cgrid(nAct), 256, 0, stream>>>(x,  Xb,  nAct);
    f32_to_bf16_kernel<<<cgrid(nW),   256, 0, stream>>>(Wq, Wqb, nW);
    f32_to_bf16_kernel<<<cgrid(nW),   256, 0, stream>>>(Wk, Wkb, nW);
    f32_to_bf16_kernel<<<cgrid(nW),   256, 0, stream>>>(Wv, Wvb, nW);
    f32_to_bf16_kernel<<<cgrid(nW),   256, 0, stream>>>(Wo, Wob, nW);

    dim3 gg(HIDDEN / 128, MTOT / 128);    // (16, 32)
    gemm_rope_kernel<MODE_Q><<<gg, 256, 0, stream>>>(Xb, Wqb, Qg);
    gemm_rope_kernel<MODE_K><<<gg, 256, 0, stream>>>(Xb, Wkb, Kg);
    gemm_rope_kernel<MODE_V><<<gg, 256, 0, stream>>>(Xb, Wvb, Vg);

    attention_kernel<<<dim3(SEQ / 64, BATCH * HEADS), 128, 0, stream>>>(Qg, Kg, Vg, AOb);

    gemm_rope_kernel<MODE_OUT><<<gg, 256, 0, stream>>>(AOb, Wob, d_out);
}
